// ResNet_imagenet_23218593202475
// MI455X (gfx1250) — 8-bit matrix operand order repaired and checked by reading, compiled, not run on hardware
//
#include <hip/hip_runtime.h>
#include <hip/hip_bf16.h>


typedef int v8i __attribute__((ext_vector_type(8)));

#define MBX(x) ((size_t)(x) * 1024u * 1024u)

static __device__ __forceinline__ signed char sgn8(float v) {
    return (v >= 0.0f) ? (signed char)1 : (signed char)-1;
}

static __device__ __forceinline__ v8i load_afrag(const signed char* p) {
    const int2* q = (const int2*)p;
    int2 x0 = q[0], x1 = q[2], x2 = q[4], x3 = q[6];
    v8i r;
    r[0] = x0.x; r[1] = x0.y; r[2] = x1.x; r[3] = x1.y;
    r[4] = x2.x; r[5] = x2.y; r[6] = x3.x; r[7] = x3.y;
    return r;
}

static __device__ __forceinline__ v8i load_bfrag(const signed char* p) {
    const int2* q = (const int2*)p;
    int2 x0 = q[0], x1 = q[2], x2 = q[4], x3 = q[6];
    v8i r;
    r[0] = x0.x; r[1] = x0.y; r[2] = x1.x; r[3] = x1.y;
    r[4] = x2.x; r[5] = x2.y; r[6] = x3.x; r[7] = x3.y;
    return r;
}

__global__ __launch_bounds__(128)
void bgemm_wmma_i8(const signed char* __restrict__ A,
                   const signed char* __restrict__ B,
                   float* __restrict__ Co,
                   int M, int N, int K, int ldc) {
    const int wave = threadIdx.x >> 5;
    const int lane = threadIdx.x & 31;
    const int mt = (M + 31) >> 5;
    const int nt = (N + 31) >> 5;
    const long tiles = (long)mt * nt;
    const long tile = (long)blockIdx.x * 4 + wave;
    if (tile >= tiles) return;
    const int tm = (int)(tile / nt);
    const int tn = (int)(tile % nt);
    const int half = lane >> 4;
    const int l15  = lane & 15;

    const long aB0 = (long)(tm * 32 + l15) * K + (half << 3);
    const long aB1 = aB0 + 16L * K;
    const long bB0 = (long)(tn * 32 + l15) * K + (half << 3);
    const long bB1 = bB0 + 16L * K;

    v8i acc00 = {0,0,0,0,0,0,0,0};
    v8i acc01 = {0,0,0,0,0,0,0,0};
    v8i acc10 = {0,0,0,0,0,0,0,0};
    v8i acc11 = {0,0,0,0,0,0,0,0};

    for (int k0 = 0; k0 < K; k0 += 64) {
        if (k0 + 64 < K) {
            __builtin_prefetch(A + aB0 + k0 + 64, 0, 3);
            __builtin_prefetch(A + aB1 + k0 + 64, 0, 3);
        }
        v8i a0 = load_afrag(A + aB0 + k0);
        v8i a1 = load_afrag(A + aB1 + k0);
        v8i b0 = load_bfrag(B + bB0 + k0);
        v8i b1 = load_bfrag(B + bB1 + k0);
        acc00 = __builtin_amdgcn_wmma_i32_16x16x64_iu8(true, a0, true, b0, acc00, false, false);
        acc01 = __builtin_amdgcn_wmma_i32_16x16x64_iu8(true, a0, true, b1, acc01, false, false);
        acc10 = __builtin_amdgcn_wmma_i32_16x16x64_iu8(true, a1, true, b0, acc10, false, false);
        acc11 = __builtin_amdgcn_wmma_i32_16x16x64_iu8(true, a1, true, b1, acc11, false, false);
    }

    const int n0 = tn * 32 + l15;
    const int n1 = n0 + 16;
#pragma unroll
    for (int r = 0; r < 8; ++r) {
        const int m0 = tm * 32 + r + half * 8;
        const int m1 = m0 + 16;
        if (m0 < M && n0 < N) Co[(long)m0 * ldc + n0] = (float)acc00[r];
        if (m0 < M && n1 < N) Co[(long)m0 * ldc + n1] = (float)acc01[r];
        if (m1 < M && n0 < N) Co[(long)m1 * ldc + n0] = (float)acc10[r];
        if (m1 < M && n1 < N) Co[(long)m1 * ldc + n1] = (float)acc11[r];
    }
}

__global__ void pack_sign_k(const float* __restrict__ w,
                            signed char* __restrict__ out, long n) {
    long i = (long)blockIdx.x * blockDim.x + threadIdx.x;
    if (i < n) out[i] = sgn8(w[i]);
}

__global__ void pack_sign_3x3_k(const float* __restrict__ w,
                                signed char* __restrict__ out, int O, int I) {
    long idx = (long)blockIdx.x * blockDim.x + threadIdx.x;
    long total = (long)O * I * 9;
    if (idx >= total) return;
    int kw = (int)(idx % 3); long t = idx / 3;
    int kh = (int)(t % 3);   t /= 3;
    int i  = (int)(t % I);
    int o  = (int)(t / I);
    out[((long)o * 9 + kh * 3 + kw) * I + i] = sgn8(w[idx]);
}

__global__ void conv1_direct_k(const float* __restrict__ x,
                               const float* __restrict__ w,
                               float* __restrict__ out) {
    long idx = (long)blockIdx.x * blockDim.x + threadIdx.x;
    if (idx >= 8L * 112 * 112 * 64) return;
    int c = (int)(idx & 63);
    long r = idx >> 6;
    int wo = (int)(r % 112); long t = r / 112;
    int ho = (int)(t % 112);
    int n  = (int)(t / 112);
    float s = 0.0f;
    for (int i = 0; i < 3; ++i)
        for (int kh = 0; kh < 7; ++kh) {
            int hi = ho * 2 - 3 + kh;
            if (hi < 0 || hi >= 224) continue;
            for (int kw = 0; kw < 7; ++kw) {
                int wi = wo * 2 - 3 + kw;
                if (wi < 0 || wi >= 224) continue;
                float xv = x[((long)(n * 3 + i) * 224 + hi) * 224 + wi];
                float wv = w[((c * 3 + i) * 7 + kh) * 7 + kw];
                s += (wv >= 0.0f) ? xv : -xv;
            }
        }
    out[idx] = s;
}

__global__ void maxpool3_k(const float* __restrict__ in, float* __restrict__ out) {
    long idx = (long)blockIdx.x * blockDim.x + threadIdx.x;
    if (idx >= 8L * 56 * 56 * 64) return;
    int c = (int)(idx & 63);
    long r = idx >> 6;
    int wo = (int)(r % 56); long t = r / 56;
    int ho = (int)(t % 56);
    int n  = (int)(t / 56);
    float m = -3.4e38f;
    for (int kh = 0; kh < 3; ++kh) {
        int hi = ho * 2 - 1 + kh;
        if (hi < 0 || hi >= 112) continue;
        for (int kw = 0; kw < 3; ++kw) {
            int wi = wo * 2 - 1 + kw;
            if (wi < 0 || wi >= 112) continue;
            float v = in[(((long)n * 112 + hi) * 112 + wi) * 64 + c];
            m = fmaxf(m, v);
        }
    }
    out[idx] = m;
}

__global__ void bn_stats_k(const float* __restrict__ x, float* __restrict__ stats,
                           long rows, int C) {
    long r0 = (long)blockIdx.x * 128;
    long r1 = r0 + 128; if (r1 > rows) r1 = rows;
    for (int c = threadIdx.x; c < C; c += 256) {
        float s = 0.0f, q = 0.0f;
        for (long r = r0; r < r1; ++r) {
            float v = x[r * C + c];
            s += v; q += v * v;
        }
        atomicAdd(&stats[c], s);
        atomicAdd(&stats[C + c], q);
    }
}

__global__ void bn_apply_k(const float* __restrict__ x, const float* __restrict__ stats,
                           const float* __restrict__ g, const float* __restrict__ b,
                           const float* __restrict__ residual,
                           float* __restrict__ outf, signed char* __restrict__ outs,
                           long rows, int C) {
    long idx = (long)blockIdx.x * blockDim.x + threadIdx.x;
    if (idx >= rows * C) return;
    int c = (int)(idx % C);
    float inv = 1.0f / (float)rows;
    float m  = stats[c] * inv;
    float var = stats[C + c] * inv - m * m;
    float sc = g[c] * rsqrtf(var + 1e-5f);
    float y = (x[idx] - m) * sc + b[c];
    if (residual) y += residual[idx];
    if (outf) outf[idx] = y;
    if (outs) outs[idx] = sgn8(y);
}

__global__ void im2col3x3_k(const signed char* __restrict__ in, signed char* __restrict__ out,
                            int H, int W, int C, int stride, int Ho, int Wo) {
    long total = (long)8 * Ho * Wo * 9 * C;
    long idx = (long)blockIdx.x * blockDim.x + threadIdx.x;
    if (idx >= total) return;
    int c = (int)(idx % C); long t = idx / C;
    int kw = (int)(t % 3); t /= 3;
    int kh = (int)(t % 3); t /= 3;
    int wo = (int)(t % Wo); t /= Wo;
    int ho = (int)(t % Ho);
    int n  = (int)(t / Ho);
    int hi = ho * stride - 1 + kh;
    int wi = wo * stride - 1 + kw;
    signed char v = 0;
    if (hi >= 0 && hi < H && wi >= 0 && wi < W)
        v = in[(((long)n * H + hi) * W + wi) * C + c];
    out[idx] = v;
}

__global__ void gather_rows_k(const signed char* __restrict__ in, signed char* __restrict__ out,
                              int H, int W, int C, int stride, int Ho, int Wo) {
    long total = (long)8 * Ho * Wo * C;
    long idx = (long)blockIdx.x * blockDim.x + threadIdx.x;
    if (idx >= total) return;
    int c = (int)(idx % C); long t = idx / C;
    int wo = (int)(t % Wo); t /= Wo;
    int ho = (int)(t % Ho);
    int n  = (int)(t / Ho);
    out[idx] = in[(((long)n * H + ho * stride) * W + wo * stride) * C + c];
}

__global__ void avgpool7_k(const float* __restrict__ in, float* __restrict__ pooled,
                           signed char* __restrict__ psign) {
    int idx = blockIdx.x * blockDim.x + threadIdx.x;
    if (idx >= 8 * 2048) return;
    int c = idx & 2047;
    int n = idx >> 11;
    float s = 0.0f;
    for (int p = 0; p < 49; ++p) s += in[((long)n * 49 + p) * 2048 + c];
    s *= (1.0f / 49.0f);
    pooled[idx] = s;
    psign[idx] = sgn8(s);
}

__global__ void bn1d_fc_k(float* __restrict__ fc, const float* __restrict__ bias,
                          const float* __restrict__ g, const float* __restrict__ b) {
    int c = blockIdx.x * blockDim.x + threadIdx.x;
    if (c >= 1000) return;
    float v[8]; float s = 0.0f;
    for (int r = 0; r < 8; ++r) { v[r] = fc[r * 1000 + c] + bias[c]; s += v[r]; }
    float m = s * 0.125f;
    float q = 0.0f;
    for (int r = 0; r < 8; ++r) { float d = v[r] - m; q += d * d; }
    float sc = g[c] * rsqrtf(q * 0.125f + 1e-5f);
    for (int r = 0; r < 8; ++r) fc[r * 1000 + c] = (v[r] - m) * sc + b[c];
}

__global__ void logsoftmax_k(const float* __restrict__ in, float* __restrict__ out) {
    __shared__ float red[256];
    const int r = blockIdx.x;
    const int tid = threadIdx.x;
    float mx = -3.4e38f;
    for (int c = tid; c < 1000; c += 256) mx = fmaxf(mx, in[r * 1000 + c]);
    red[tid] = mx; __syncthreads();
    for (int s2 = 128; s2 > 0; s2 >>= 1) {
        if (tid < s2) red[tid] = fmaxf(red[tid], red[tid + s2]);
        __syncthreads();
    }
    mx = red[0]; __syncthreads();
    float s = 0.0f;
    for (int c = tid; c < 1000; c += 256) s += expf(in[r * 1000 + c] - mx);
    red[tid] = s; __syncthreads();
    for (int s2 = 128; s2 > 0; s2 >>= 1) {
        if (tid < s2) red[tid] += red[tid + s2];
        __syncthreads();
    }
    float lse = logf(red[0]) + mx;
    for (int c = tid; c < 1000; c += 256) out[r * 1000 + c] = in[r * 1000 + c] - lse;
}

static inline unsigned gcnt(long n, int b) { return (unsigned)((n + b - 1) / b); }

static void launch_bgemm(const signed char* A, const signed char* B, float* C,
                         int M, int N, int K, int ldc, hipStream_t stream) {
    long tiles = (long)((M + 31) / 32) * ((N + 31) / 32);
    bgemm_wmma_i8<<<gcnt(tiles, 4), 128, 0, stream>>>(A, B, C, M, N, K, ldc);
}

static void run_bn(const float* xin, float* stats, const float* g, const float* b,
                   const float* res, float* outf, signed char* outs,
                   long rows, int C, hipStream_t stream) {
    hipMemsetAsync(stats, 0, 2 * (size_t)C * sizeof(float), stream);
    bn_stats_k<<<gcnt(rows, 128), 256, 0, stream>>>(xin, stats, rows, C);
    bn_apply_k<<<gcnt(rows * (long)C, 256), 256, 0, stream>>>(xin, stats, g, b, res,
                                                              outf, outs, rows, C);
}

extern "C" void kernel_launch(void* const* d_in, const int* in_sizes, int n_in,
                              void* d_out, int out_size, void* d_ws, size_t ws_size,
                              hipStream_t stream) {
    (void)in_sizes; (void)n_in; (void)out_size; (void)ws_size;

    int pi = 0;
    const float* x       = (const float*)d_in[pi++];
    const float* conv1_w = (const float*)d_in[pi++];
    const float* bn1_g   = (const float*)d_in[pi++];
    const float* bn1_b   = (const float*)d_in[pi++];

    char* ws = (char*)d_ws;
    float*       X    = (float*)(ws + 0);
    float*       X2   = (float*)(ws + MBX(32));
    signed char* Xb   = (signed char*)(ws + MBX(64));
    signed char* Xb2  = (signed char*)(ws + MBX(72));
    float*       T    = (float*)(ws + MBX(80));
    float*       Y    = (float*)(ws + MBX(96));
    float*       R    = (float*)(ws + MBX(128));
    signed char* Im   = (signed char*)(ws + MBX(160));
    signed char* S    = (signed char*)(ws + MBX(176));
    signed char* Wp   = (signed char*)(ws + MBX(184));
    float*       stats  = (float*)(ws + MBX(188));
    float*       stats2 = stats + 4096;
    float*       pooled = stats2 + 4096;
    signed char* psign  = (signed char*)(pooled + 8 * 2048);
    float*       fcout  = (float*)(psign + 32 * 2048);

    conv1_direct_k<<<gcnt(8L * 112 * 112 * 64, 256), 256, 0, stream>>>(x, conv1_w, Y);
    maxpool3_k<<<gcnt(8L * 56 * 56 * 64, 256), 256, 0, stream>>>(Y, X);
    run_bn(X, stats, bn1_g, bn1_b, nullptr, X, Xb, 8L * 56 * 56, 64, stream);

    const int LAYERS_[4]  = {3, 4, 23, 3};
    const int PLANES_[4]  = {64, 128, 256, 512};
    const int STRIDES_[4] = {1, 2, 2, 2};
    int H = 56, W = 56, Cin = 64;
    float* cur = X;  signed char* curS = Xb;
    float* alt = X2; signed char* altS = Xb2;

    for (int st = 0; st < 4; ++st) {
        const int planes = PLANES_[st];
        const int Cout = planes * 4;
        for (int blk = 0; blk < LAYERS_[st]; ++blk) {
            const float* c1w = (const float*)d_in[pi++];
            const float* g1  = (const float*)d_in[pi++];
            const float* b1  = (const float*)d_in[pi++];
            const float* c2w = (const float*)d_in[pi++];
            const float* g2  = (const float*)d_in[pi++];
            const float* b2  = (const float*)d_in[pi++];
            const float* c3w = (const float*)d_in[pi++];
            const float* g3  = (const float*)d_in[pi++];
            const float* b3  = (const float*)d_in[pi++];
            const bool ds = (blk == 0);
            const float *cdw = nullptr, *gd = nullptr, *bd = nullptr;
            if (ds) {
                cdw = (const float*)d_in[pi++];
                gd  = (const float*)d_in[pi++];
                bd  = (const float*)d_in[pi++];
            }
            const int s = (blk == 0) ? STRIDES_[st] : 1;
            const int Ho = H / s, Wo = W / s;
            const long rin = 8L * H * W, rout = 8L * Ho * Wo;

            pack_sign_k<<<gcnt((long)planes * Cin, 256), 256, 0, stream>>>(c1w, Wp, (long)planes * Cin);
            launch_bgemm(curS, Wp, T, (int)rin, planes, Cin, planes, stream);
            run_bn(T, stats, g1, b1, nullptr, nullptr, S, rin, planes, stream);

            im2col3x3_k<<<gcnt(rout * 9 * planes, 256), 256, 0, stream>>>(S, Im, H, W, planes, s, Ho, Wo);
            pack_sign_3x3_k<<<gcnt((long)planes * planes * 9, 256), 256, 0, stream>>>(c2w, Wp, planes, planes);
            launch_bgemm(Im, Wp, T, (int)rout, planes, 9 * planes, planes, stream);
            run_bn(T, stats, g2, b2, nullptr, nullptr, S, rout, planes, stream);

            pack_sign_k<<<gcnt((long)Cout * planes, 256), 256, 0, stream>>>(c3w, Wp, (long)Cout * planes);
            launch_bgemm(S, Wp, Y, (int)rout, Cout, planes, Cout, stream);

            const float* res;
            if (ds) {
                pack_sign_k<<<gcnt((long)Cout * Cin, 256), 256, 0, stream>>>(cdw, Wp, (long)Cout * Cin);
                const signed char* Ad = curS;
                if (s != 1) {
                    gather_rows_k<<<gcnt(rout * Cin, 256), 256, 0, stream>>>(curS, Im, H, W, Cin, s, Ho, Wo);
                    Ad = Im;
                }
                launch_bgemm(Ad, Wp, R, (int)rout, Cout, Cin, Cout, stream);
                run_bn(R, stats2, gd, bd, nullptr, R, nullptr, rout, Cout, stream);
                res = R;
            } else {
                res = cur;
            }

            run_bn(Y, stats, g3, b3, res, alt, altS, rout, Cout, stream);

            { float* tf = cur; cur = alt; alt = tf; }
            { signed char* ts = curS; curS = altS; altS = ts; }
            H = Ho; W = Wo; Cin = Cout;
        }
    }

    const float* fc_w  = (const float*)d_in[pi++];
    const float* fc_b  = (const float*)d_in[pi++];
    const float* bnf_g = (const float*)d_in[pi++];
    const float* bnf_b = (const float*)d_in[pi++];

    hipMemsetAsync(psign, 0, (size_t)32 * 2048, stream);
    hipMemsetAsync(Wp, 0, (size_t)1024 * 2048, stream);
    avgpool7_k<<<gcnt(8 * 2048, 256), 256, 0, stream>>>(cur, pooled, psign);
    pack_sign_k<<<gcnt(1000L * 2048, 256), 256, 0, stream>>>(fc_w, Wp, 1000L * 2048);
    launch_bgemm(psign, Wp, fcout, 8, 1000, 2048, 1000, stream);
    bn1d_fc_k<<<gcnt(1000, 256), 256, 0, stream>>>(fcout, fc_b, bnf_g, bnf_b);
    logsoftmax_k<<<8, 256, 0, stream>>>(fcout, (float*)d_out);
}
